// LSTMModel_12936441495768
// MI455X (gfx1250) — matrix operand order repaired and checked by reading, not compiled, not run on hardware
//
#include <hip/hip_runtime.h>

// 10-layer stacked LSTM, B=256 T=512 IN=34 H=80 NC=5, for MI455X (gfx1250).
// Batch-sliced WMMA scan: 16 batch slices x 16 rows, 5 waves/WG, per wave 4
// gate tiles (i,f,g,o) 16x16, K=160 fused [x|h] via v_wmma_f32_16x16x32_f16.
// B operands + biases live in VGPRs for the whole 512-step scan; the shared
// A panel (x_t|h_{t-1}) is double-buffered in LDS -> one s_barrier per step.
// Gates use v_tanh_f32 (single TRANS32 op each).
//
// Two modes, chosen by ws_size at launch:
//  - PIPELINED (210MB scratch): grid 16x10, one WG per (slice, layer).
//    Layer l consumes layer l-1's per-timestep output via acquire-polled
//    progress counters; release-published one step late so the producer's
//    store fence is always already satisfied. Critical path ~T+2L steps
//    instead of T*L (~10x latency reduction).
//  - SEQUENTIAL (42MB scratch): grid 16, each WG runs all 10 layers with
//    f16 activation ping-pong in scratch.

#define B_    256
#define T_    512
#define IN_   34
#define H_    80
#define L_    10
#define NC_   5
#define G_    (4 * H_)       // 320 gate outputs
#define KX_   80             // x half of K (padded from 34 for layer 0)
#define KTOT_ 160            // [x | h] fused K
#define THREADS_ 160         // 5 waves of 32
#define APANEL_ (16 * KTOT_) // one A buffer: 16 rows x 160 halfs
#define BTH_  ((size_t)B_ * T_ * H_)   // elements of one activation buffer

typedef __attribute__((ext_vector_type(16))) _Float16 v16h;
typedef __attribute__((ext_vector_type(8)))  _Float16 v8h;
typedef __attribute__((ext_vector_type(8)))  float    v8f;

__device__ __forceinline__ float ftanh(float x) {
#if __has_builtin(__builtin_amdgcn_tanhf)
    return __builtin_amdgcn_tanhf(x);       // v_tanh_f32 (TRANS32)
#else
    float e = __expf(2.0f * x);
    return (e - 1.0f) / (e + 1.0f);
#endif
}
__device__ __forceinline__ float fsigmoid(float x) {
    return 0.5f * ftanh(0.5f * x) + 0.5f;   // 1 trans op per gate
}

// Fused weight element W[g][k] of [Wih | Whh], k in [0,160), x part zero-padded.
__device__ __forceinline__ float wfetch(const float* __restrict__ wih, int kin,
                                        const float* __restrict__ whh,
                                        int g, int k) {
    if (k < KX_) return (k < kin) ? wih[g * kin + k] : 0.0f;
    return whh[g * H_ + (k - KX_)];
}

// ---- prologue: pad/convert x [B,T,34] f32 -> [B,T,80] f16 ----
__global__ __launch_bounds__(256)
void pad_x_f16(const float* __restrict__ x, _Float16* __restrict__ xp) {
    size_t i = (size_t)blockIdx.x * blockDim.x + threadIdx.x;   // over B*T*H
    if (i >= BTH_) return;
    int k = (int)(i % H_);
    size_t bt = i / H_;
    xp[i] = (_Float16)((k < IN_) ? x[bt * IN_ + k] : 0.0f);
}

// One layer's 512-step scan for one 16-row batch slice.
// flagIn  != null: poll upstream progress (acquire) before each timestep.
// flagOut != null: publish own progress (release), one step deferred.
// dst     != null: store h_t f16 to the next layer's activation buffer.
__device__ __forceinline__ void scan_layer(
    int l, int b0, int tid, int wv, int nloc, bool hiH, int mb,
    int xr, int xq, int nh,
    const _Float16* __restrict__ src, _Float16* __restrict__ dst,
    int* flagIn, int* flagOut,
    const float* __restrict__ Wih0, const float* __restrict__ WihR,
    const float* __restrict__ Whh,  const float* __restrict__ bih,
    const float* __restrict__ bhh,  _Float16* Alds)
{
    const float* whh_l = Whh + (size_t)l * G_ * H_;
    const float* wih_l = (l == 0) ? Wih0 : (WihR + (size_t)(l - 1) * G_ * H_);
    const int kin = (l == 0) ? IN_ : H_;

    // ---- per-layer register state: B fragments (4x5 v16h = 160 VGPRs) ----
    // B layout: lane holds column g, 16 contiguous K per k-step (lo/hi half).
    v16h bfr[4][5];
    #pragma unroll
    for (int gti = 0; gti < 4; ++gti) {
        const int g = (wv + 5 * gti) * 16 + nloc;
        #pragma unroll
        for (int s5 = 0; s5 < 5; ++s5) {
            const int kb = s5 * 32 + (hiH ? 16 : 0);
            v16h bf;
            #pragma unroll
            for (int kk = 0; kk < 16; ++kk)
                bf[kk] = (_Float16)wfetch(wih_l, kin, whh_l, g, kb + kk);
            bfr[gti][s5] = bf;
        }
    }
    const float bI = bih[l * G_ + (wv     ) * 16 + nloc] + bhh[l * G_ + (wv     ) * 16 + nloc];
    const float bF = bih[l * G_ + (wv + 5 ) * 16 + nloc] + bhh[l * G_ + (wv + 5 ) * 16 + nloc];
    const float bG = bih[l * G_ + (wv + 10) * 16 + nloc] + bhh[l * G_ + (wv + 10) * 16 + nloc];
    const float bO = bih[l * G_ + (wv + 15) * 16 + nloc] + bhh[l * G_ + (wv + 15) * 16 + nloc];
    v8f c_acc = {};                            // c_0 = 0

    // h_0 = 0 in panel 0's h-half
    for (int e = tid; e < 16 * H_; e += THREADS_) {
        int m = e / H_, k = e - m * H_;
        Alds[m * KTOT_ + KX_ + k] = (_Float16)0.0f;
    }
    __syncthreads();

    for (int t = 0; t < T_; ++t) {
        _Float16* bufR = Alds + (t & 1) * APANEL_;         // read panel
        _Float16* bufN = Alds + ((t + 1) & 1) * APANEL_;   // next panel

        // ---- wait for upstream layer to publish timestep t ----
        if (flagIn) {
            while (__hip_atomic_load(flagIn, __ATOMIC_ACQUIRE,
                                     __HIP_MEMORY_SCOPE_AGENT) <= t)
                __builtin_amdgcn_s_sleep(2);
            // acquire also invalidates L0 -> no stale lines from neighbors
        }

        // ---- x_t into read panel (one 16B vector load per thread) ----
        const _Float16* sp = src + ((size_t)(b0 + xr) * T_ + t) * H_ + xq * 8;
        *(v8h*)(bufR + xr * KTOT_ + xq * 8) = *(const v8h*)sp;
        if (!flagIn && t + 1 < T_)
            __builtin_prefetch(sp + H_, 0, 0);             // global_prefetch_b8

        if (flagOut) __threadfence();   // h stores of step t-1 (a full step old)
        __syncthreads();                // x_t + h_{t-1} visible in LDS
        if (flagOut && tid == 0 && t > 0)
            __hip_atomic_store(flagOut, t, __ATOMIC_RELEASE,
                               __HIP_MEMORY_SCOPE_AGENT);  // publish t steps done

        // ---- A fragments, contiguous K (16 per lane half, matches B): 5 k-steps of 32 ----
        v16h afrag[5];
        {
            const _Float16* arow = bufR + nloc * KTOT_;
            const int ko = hiH ? 16 : 0;
            #pragma unroll
            for (int s5 = 0; s5 < 5; ++s5) {
                v8h a0 = *(const v8h*)(arow + s5 * 32 + ko);
                v8h a1 = *(const v8h*)(arow + s5 * 32 + 8 + ko);
                afrag[s5] = __builtin_shufflevector(a0, a1,
                    0,1,2,3,4,5,6,7,8,9,10,11,12,13,14,15);
            }
        }

        // ---- 4 gate tiles x 5 k-steps, B from registers ----
        v8f gacc[4];
        #pragma unroll
        for (int gti = 0; gti < 4; ++gti) {
            v8f acc = {};
            #pragma unroll
            for (int s5 = 0; s5 < 5; ++s5)
                acc = __builtin_amdgcn_wmma_f32_16x16x32_f16(
                    false, afrag[s5], false, bfr[gti][s5],
                    (short)0, acc, false, false);
            gacc[gti] = acc;
        }

        // ---- gates / cell update (f32); h_t -> next panel (+ scratch) ----
        #pragma unroll
        for (int v = 0; v < 8; ++v) {
            float iv = fsigmoid(gacc[0][v] + bI);
            float fv = fsigmoid(gacc[1][v] + bF);
            float gv = ftanh(gacc[2][v] + bG);
            float ov = fsigmoid(gacc[3][v] + bO);
            float cv = fv * c_acc[v] + iv * gv;
            c_acc[v] = cv;
            _Float16 hv = (_Float16)(ov * ftanh(cv));
            bufN[(mb + v) * KTOT_ + KX_ + nh] = hv;        // feed t+1
            if (dst)
                dst[((size_t)(b0 + mb + v) * T_ + t) * H_ + nh] = hv;
        }
    }

    if (flagOut) {   // final publish: all T steps done
        __threadfence();
        __syncthreads();
        if (tid == 0)
            __hip_atomic_store(flagOut, T_, __ATOMIC_RELEASE,
                               __HIP_MEMORY_SCOPE_AGENT);
    }
}

__device__ __forceinline__ void fc_tail(const _Float16* Alds, int b0, int tid,
                                        const float* __restrict__ fcw,
                                        const float* __restrict__ fcb,
                                        float* __restrict__ out)
{
    // h_{T-1}: T even -> lives in panel 0's h-half
    if (tid < 16 * NC_) {
        int m = tid / NC_, cc = tid - m * NC_;
        float acc = fcb[cc];
        #pragma unroll 4
        for (int k = 0; k < H_; ++k)
            acc += (float)Alds[m * KTOT_ + KX_ + k] * fcw[cc * H_ + k];
        out[(b0 + m) * NC_ + cc] = acc;
    }
}

template <bool PIPE>
__global__ __launch_bounds__(THREADS_, 1)
void lstm10_wmma(const float* __restrict__ Wih0,   // [4H,IN]
                 const float* __restrict__ WihR,   // [L-1,4H,H]
                 const float* __restrict__ Whh,    // [L,4H,H]
                 const float* __restrict__ bih,    // [L,4H]
                 const float* __restrict__ bhh,    // [L,4H]
                 const float* __restrict__ fcw,    // [NC,H]
                 const float* __restrict__ fcb,    // [NC]
                 float* __restrict__ out,          // [B,NC]
                 _Float16* __restrict__ act,       // activation buffers (f16)
                 int* __restrict__ flags)          // [16][L-1] progress (PIPE)
{
    __shared__ _Float16 Alds[2 * APANEL_];         // double-buffered [x_t | h_{t-1}]

    const int tid   = threadIdx.x;
    const int wv    = tid >> 5;                    // wave 0..4 -> hidden cols [16wv,16wv+16)
    const int lane  = tid & 31;
    const int nloc  = lane & 15;                   // A row / B,C column within tile
    const bool hiH  = lane >= 16;
    const int  mb   = hiH ? 8 : 0;                 // C-layout row base
    const int  b0   = blockIdx.x * 16;             // batch slice base
    const int  xr   = tid / 10;                    // cooperative A-panel row
    const int  xq   = tid - xr * 10;               // 10 chunks of 8 halfs per row
    const int  nh   = wv * 16 + nloc;              // this lane's hidden column

    if constexpr (PIPE) {
        // one WG per (slice, layer); wavefront over timesteps
        const int l = blockIdx.y;
        const _Float16* src = act + (size_t)l * BTH_;
        _Float16* dst = (l < L_ - 1) ? (act + (size_t)(l + 1) * BTH_) : nullptr;
        int* fin  = (l > 0)      ? &flags[blockIdx.x * (L_ - 1) + (l - 1)] : nullptr;
        int* fout = (l < L_ - 1) ? &flags[blockIdx.x * (L_ - 1) + l]       : nullptr;
        scan_layer(l, b0, tid, wv, nloc, hiH, mb, xr, xq, nh,
                   src, dst, fin, fout, Wih0, WihR, Whh, bih, bhh, Alds);
        if (l == L_ - 1)
            fc_tail(Alds, b0, tid, fcw, fcb, out);
    } else {
        // all layers sequentially in one WG; ping-pong activations
        const _Float16* src = act + BTH_;          // x-pad (ping)
        _Float16*       dst = act;                 // pong
        for (int l = 0; l < L_; ++l) {
            scan_layer(l, b0, tid, wv, nloc, hiH, mb, xr, xq, nh,
                       src, (l == L_ - 1) ? nullptr : dst,
                       nullptr, nullptr, Wih0, WihR, Whh, bih, bhh, Alds);
            __threadfence();
            __syncthreads();
            const _Float16* t_ = src;
            src = dst;
            dst = (_Float16*)t_;
        }
        fc_tail(Alds, b0, tid, fcw, fcb, out);
    }
}

extern "C" void kernel_launch(void* const* d_in, const int* in_sizes, int n_in,
                              void* d_out, int out_size, void* d_ws, size_t ws_size,
                              hipStream_t stream) {
    const float* x    = (const float*)d_in[0];
    const float* Wih0 = (const float*)d_in[1];
    const float* WihR = (const float*)d_in[2];
    const float* Whh  = (const float*)d_in[3];
    const float* bih  = (const float*)d_in[4];
    const float* bhh  = (const float*)d_in[5];
    const float* fcw  = (const float*)d_in[6];
    const float* fcb  = (const float*)d_in[7];
    float* out = (float*)d_out;

    _Float16* act = (_Float16*)d_ws;
    const size_t padGrid = (BTH_ + 255) / 256;

    const size_t pipeBytes = (size_t)L_ * BTH_ * sizeof(_Float16)
                           + (size_t)(B_ / 16) * (L_ - 1) * sizeof(int);

    if (ws_size >= pipeBytes) {
        // PIPELINED: buffers 0..9 (buf0 = padded x), flags after them
        int* flags = (int*)(act + (size_t)L_ * BTH_);
        (void)hipMemsetAsync(flags, 0,
                             (size_t)(B_ / 16) * (L_ - 1) * sizeof(int), stream);
        pad_x_f16<<<(unsigned)padGrid, 256, 0, stream>>>(x, act);
        dim3 grid(B_ / 16, L_);
        lstm10_wmma<true><<<grid, THREADS_, 0, stream>>>(
            Wih0, WihR, Whh, bih, bhh, fcw, fcb, out, act, flags);
    } else {
        // SEQUENTIAL fallback: 2 ping-pong buffers (x-pad in the second)
        pad_x_f16<<<(unsigned)padGrid, 256, 0, stream>>>(x, act + BTH_);
        lstm10_wmma<false><<<dim3(B_ / 16, 1), THREADS_, 0, stream>>>(
            Wih0, WihR, Whh, bih, bhh, fcw, fcb, out, act, nullptr);
    }
}
